// Mixer2dTriU_52553219834527
// MI455X (gfx1250) — matrix operand order repaired and checked by reading, not compiled, not run on hardware
//
#include <hip/hip_runtime.h>

typedef __attribute__((ext_vector_type(16))) __bf16 v16bf;
typedef __attribute__((ext_vector_type(8)))  __bf16 v8bf;
typedef __attribute__((ext_vector_type(4)))  __bf16 v4bf;
typedef __attribute__((ext_vector_type(8)))  float  v8f;
typedef __attribute__((ext_vector_type(4)))  int    v4i;

typedef __attribute__((address_space(1))) void as1_void;
typedef __attribute__((address_space(3))) void as3_void;
typedef __attribute__((address_space(1))) v4i  as1_v4i;
typedef __attribute__((address_space(3))) v4i  as3_v4i;

#if defined(__gfx1250__) && __has_builtin(__builtin_amdgcn_global_load_async_to_lds_b128)
#define USE_ASYNC_LDS 1
#else
#define USE_ASYNC_LDS 0
#endif

namespace {

constexpr int Tdim = 64;
constexpr int Cdim = 128;
constexpr int TCn  = Tdim * Cdim;   // 8192 elements per batch tile
constexpr int LDH  = 136;           // bf16 row stride for A-side tiles (mult of 8)
constexpr int LDW  = 72;            // bf16 row stride for 64x64 tril weights

// Packed-weights image layout (bf16 elements), shared by d_ws and LDS region.
constexpr int WS_W1   = 0;                  // 32 frags x 512
constexpr int WS_W2   = WS_W1 + 32 * 512;   // 32 frags x 512
constexpr int WS_TRI  = WS_W2 + 32 * 512;   // 64 x LDW row-major
constexpr int WS_PAD  = WS_TRI + Tdim * LDW;
constexpr int WS_ELEM = 38912;              // pad to 77824 B = 19 * 256 lanes * 16 B
constexpr int WS_BYTES = WS_ELEM * 2;
constexpr int ACPY_ITERS = WS_BYTES / (256 * 16);   // 19

struct __align__(16) Smem {
  float  f[TCn];              // f32 tile: inputs -> (tm + inputs) -> x2
  __bf16 xh[Tdim * LDH];      // stage1: x post-LN1 (B-swizzled); stage2: x2 (A row-major)
  __bf16 hh[Tdim * LDH];      // hidden tile post-GELU, A row-major
  // ---- contiguous async-copied region; must match ws image exactly ----
  __bf16 w1s[32 * 512];       // w1^T, B-fragment-swizzled (K=c, N=h)
  __bf16 w2s[32 * 512];       // w2^T, B-fragment-swizzled (K=h, N=c)
  __bf16 wtri[Tdim * LDW];    // tril(triu_w), A row-major
  __bf16 wpad[WS_ELEM - WS_PAD];
  // --------------------------------------------------------------------
  float  tb[Tdim];            // triu_b
  float  bb1[Cdim];           // b1
  float  bb2[Cdim];           // b2
  float  red[18];             // cross-wave reduction scratch
};

#if USE_ASYNC_LDS
template <int I>
__device__ __forceinline__ void asyncCopyAll(as1_v4i* g, as3_v4i* l) {
  if constexpr (I < ACPY_ITERS) {
    __builtin_amdgcn_global_load_async_to_lds_b128(g, l, I * 4096, 0);
    asyncCopyAll<I + 1>(g, l);
  }
}
#endif

__device__ __forceinline__ v8f vzero8() {
  v8f z;
#pragma unroll
  for (int i = 0; i < 8; ++i) z[i] = 0.0f;
  return z;
}

__device__ __forceinline__ v16bf cat16(v8bf lo, v8bf hi) {
  return __builtin_shufflevector(lo, hi, 0, 1, 2, 3, 4, 5, 6, 7,
                                         8, 9, 10, 11, 12, 13, 14, 15);
}

// Element (k, n) -> offset in B-fragment-swizzled buffer with `ntiles` N tiles.
// Fragment tile = 32(K) x 16(N) = 512 bf16; lane = (n%16) | ((k/16)%2)<<4;
// element e = k%16; each lane's 16 bf16 are contiguous (32B).
__device__ __forceinline__ int bswz(int k, int n, int ntiles) {
  return (((k >> 5) * ntiles + (n >> 4)) << 9) +
         ((((n & 15) | (((k >> 4) & 1) << 4))) << 4) + (k & 15);
}

// A fragment (16x32 bf16, M x K) from a row-major tile: two 16B LDS loads.
__device__ __forceinline__ v16bf loadA(const __bf16* s, int lda, int m0, int k0, int lane) {
  const int m  = m0 + (lane & 15);
  const int hb = (lane >> 4) & 1;
  const v8bf* p = reinterpret_cast<const v8bf*>(s + m * lda + k0 + hb * 16);
  return cat16(p[0], p[1]);   // K = k0+hb*16+0..15
}

// B fragment (32x16 bf16, K x N) from a swizzled buffer: two 16B LDS loads.
__device__ __forceinline__ v16bf loadBs(const __bf16* s, int ntiles, int kt, int nt, int lane) {
  const v8bf* p = reinterpret_cast<const v8bf*>(
      s + (((kt * ntiles + nt) << 9) + (lane << 4)));
  return cat16(p[0], p[1]);
}

__device__ __forceinline__ v8f wmma_bf16(v16bf a, v16bf b, v8f c) {
  return __builtin_amdgcn_wmma_f32_16x16x32_bf16(false, a, false, b, (short)0, c,
                                                 false, false);
}

// Reduce two values across the 256-thread block (8 waves of 32).
__device__ __forceinline__ void blockReduce2(float& a, float& b, float* red, int tid) {
#pragma unroll
  for (int o = 16; o > 0; o >>= 1) {
    a += __shfl_down(a, o, 32);
    b += __shfl_down(b, o, 32);
  }
  const int w = tid >> 5;
  if ((tid & 31) == 0) { red[w] = a; red[8 + w] = b; }
  __syncthreads();
  if (tid == 0) {
    float sa = 0.f, sb = 0.f;
#pragma unroll
    for (int i = 0; i < 8; ++i) { sa += red[i]; sb += red[8 + i]; }
    red[16] = sa; red[17] = sb;
  }
  __syncthreads();
  a = red[16];
  b = red[17];
  __syncthreads();  // protect red[] before reuse
}

// ---- One-shot weight packing: f32 -> bf16, WMMA-swizzled, into d_ws ----
__global__ __launch_bounds__(256)
void prep_weights_kernel(const float* __restrict__ triw,
                         const float* __restrict__ w1,
                         const float* __restrict__ w2,
                         __bf16* __restrict__ wsb) {
  const int t = blockIdx.x * 256 + threadIdx.x;   // 0..4095 (float4 units)
  {
    int idx = t << 2;
    int r = idx >> 7;               // first index  (= n)
    int c = idx & 127;              // second index (= k), 4-aligned
    float4 a  = reinterpret_cast<const float4*>(w1)[t];
    float4 bq = reinterpret_cast<const float4*>(w2)[t];
    v4bf pa = { (__bf16)a.x,  (__bf16)a.y,  (__bf16)a.z,  (__bf16)a.w  };
    v4bf pb = { (__bf16)bq.x, (__bf16)bq.y, (__bf16)bq.z, (__bf16)bq.w };
    int off = bswz(c, r, 8);
    *reinterpret_cast<v4bf*>(wsb + WS_W1 + off) = pa;
    *reinterpret_cast<v4bf*>(wsb + WS_W2 + off) = pb;
  }
  if (t < 1024) {                   // 64x64 tril weights
    int idx = t << 2;
    int tt = idx >> 6, jj = idx & 63;   // jj 4-aligned
    float4 a = reinterpret_cast<const float4*>(triw)[t];
    v4bf p = { (__bf16)((jj + 0 <= tt) ? a.x : 0.0f),
               (__bf16)((jj + 1 <= tt) ? a.y : 0.0f),
               (__bf16)((jj + 2 <= tt) ? a.z : 0.0f),
               (__bf16)((jj + 3 <= tt) ? a.w : 0.0f) };
    *reinterpret_cast<v4bf*>(wsb + WS_TRI + tt * LDW + jj) = p;
  }
}

__global__ __launch_bounds__(256)
void mixer2d_triu_kernel(const float* __restrict__ inp,
                         const float* __restrict__ ln1g, const float* __restrict__ ln1b,
                         const float* __restrict__ ln2g, const float* __restrict__ ln2b,
                         const float* __restrict__ triw, const float* __restrict__ trib,
                         const float* __restrict__ w1,   const float* __restrict__ b1,
                         const float* __restrict__ w2,   const float* __restrict__ b2,
                         const __bf16* __restrict__ wsb,
                         float* __restrict__ out) {
  extern __shared__ unsigned char smraw[];
  Smem& sm = *reinterpret_cast<Smem*>(smraw);

  const int tid  = threadIdx.x;
  const int lane = tid & 31;
  const int wave = tid >> 5;
  const int bidx = blockIdx.x;
  const float* gin = inp + (size_t)bidx * TCn;

#if USE_ASYNC_LDS
  // ---- Kick off async DMA of the packed-weight image into LDS ----
  {
    const unsigned char* gsrc_b = (const unsigned char*)wsb + tid * 16;
    unsigned char*       ldst_b = (unsigned char*)&sm.w1s[0] + tid * 16;
    as1_v4i* gsrc = (as1_v4i*)(as1_void*)gsrc_b;
    as3_v4i* ldst = (as3_v4i*)(as3_void*)ldst_b;
    asyncCopyAll<0>(gsrc, ldst);
  }
#endif

  // ---- Stage inputs (one HBM pass) + partial sums for LN1 ----
  float s1 = 0.f, s2 = 0.f;
  float4 v[8];
  const float4* gin4 = reinterpret_cast<const float4*>(gin);
#pragma unroll
  for (int i = 0; i < 8; ++i) {
    float4 t = gin4[tid + i * 256];
    v[i] = t;
    s1 += t.x + t.y + t.z + t.w;
    s2 += t.x * t.x + t.y * t.y + t.z * t.z + t.w * t.w;
    reinterpret_cast<float4*>(sm.f)[tid + i * 256] = t;
  }

#if !USE_ASYNC_LDS
  // ---- Fallback: stage weights in-kernel ----
#pragma unroll
  for (int i = 0; i < 4; ++i) {
    int i4  = tid + i * 256;
    int idx = i4 << 2;
    int tt = idx >> 6, jj = idx & 63;
    float4 a = reinterpret_cast<const float4*>(triw)[i4];
    v4bf p = { (__bf16)((jj + 0 <= tt) ? a.x : 0.0f),
               (__bf16)((jj + 1 <= tt) ? a.y : 0.0f),
               (__bf16)((jj + 2 <= tt) ? a.z : 0.0f),
               (__bf16)((jj + 3 <= tt) ? a.w : 0.0f) };
    *reinterpret_cast<v4bf*>(&sm.wtri[tt * LDW + jj]) = p;
  }
#pragma unroll
  for (int i = 0; i < 16; ++i) {
    int i4  = tid + i * 256;
    int idx = i4 << 2;
    int r = idx >> 7;
    int c = idx & 127;
    float4 a  = reinterpret_cast<const float4*>(w1)[i4];
    float4 bq = reinterpret_cast<const float4*>(w2)[i4];
    v4bf pa = { (__bf16)a.x,  (__bf16)a.y,  (__bf16)a.z,  (__bf16)a.w  };
    v4bf pb = { (__bf16)bq.x, (__bf16)bq.y, (__bf16)bq.z, (__bf16)bq.w };
    int off = bswz(c, r, 8);
    *reinterpret_cast<v4bf*>(&sm.w1s[off]) = pa;
    *reinterpret_cast<v4bf*>(&sm.w2s[off]) = pb;
  }
#else
  (void)triw; (void)w1; (void)w2;
#endif
  if (tid < Tdim) sm.tb[tid]  = trib[tid];
  if (tid < Cdim) sm.bb1[tid] = b1[tid];
  if (tid < Cdim) sm.bb2[tid] = b2[tid];

  // ---- LN1 (joint over T*C) ----
  blockReduce2(s1, s2, sm.red, tid);
  const float inv = 1.0f / (float)TCn;
  float mu   = s1 * inv;
  float var  = s2 * inv - mu * mu;
  float rstd = rsqrtf(var + 1e-5f);
  // x (post-LN1) is consumed only as a WMMA B operand -> write B-swizzled.
#pragma unroll
  for (int i = 0; i < 8; ++i) {
    int i4 = tid + i * 256;
    int e0 = i4 * 4;
    float4 g4  = reinterpret_cast<const float4*>(ln1g)[i4];
    float4 be4 = reinterpret_cast<const float4*>(ln1b)[i4];
    float4 t = v[i];
    int row = e0 >> 7, col = e0 & 127;   // row = k, col = n (4-aligned)
    int off = bswz(row, col, 8);         // consecutive n -> +16 elements
    sm.xh[off     ] = (__bf16)((t.x - mu) * rstd * g4.x + be4.x);
    sm.xh[off + 16] = (__bf16)((t.y - mu) * rstd * g4.y + be4.y);
    sm.xh[off + 32] = (__bf16)((t.z - mu) * rstd * g4.z + be4.z);
    sm.xh[off + 48] = (__bf16)((t.w - mu) * rstd * g4.w + be4.w);
  }
#if USE_ASYNC_LDS
#if __has_builtin(__builtin_amdgcn_s_wait_asynccnt)
  __builtin_amdgcn_s_wait_asynccnt(0);
#else
  asm volatile("s_wait_asynccnt 0" ::: "memory");
#endif
#endif
  __syncthreads();

  // ---- tm = tril(Wtri)(64x64) x x(64x128), + triu_b + inputs -> sm.f ----
  const int nt = wave;              // N tile column (8 waves x 16 cols)
  const int hb = lane >> 4;
  const int nn = lane & 15;
  v8f acc[4];
#pragma unroll
  for (int mt = 0; mt < 4; ++mt) acc[mt] = vzero8();
#pragma unroll
  for (int ks = 0; ks < 2; ++ks) {  // K = 64
    v16bf bf = loadBs(sm.xh, 8, ks, nt, lane);
#pragma unroll
    for (int mt = 0; mt < 4; ++mt) {
      v16bf af = loadA(sm.wtri, LDW, mt * 16, ks * 32, lane);
      acc[mt] = wmma_bf16(af, bf, acc[mt]);
    }
  }
#pragma unroll
  for (int mt = 0; mt < 4; ++mt) {
#pragma unroll
    for (int r = 0; r < 8; ++r) {
      int tt = mt * 16 + hb * 8 + r;
      int cc = nt * 16 + nn;
      sm.f[tt * Cdim + cc] = acc[mt][r] + sm.tb[tt] + sm.f[tt * Cdim + cc];
    }
  }
  __syncthreads();

  // ---- LN2 (joint) over (tm + inputs) -> x2 (f32 in sm.f, bf16 A-tile in sm.xh) ----
  s1 = 0.f; s2 = 0.f;
  float4 u[8];
#pragma unroll
  for (int i = 0; i < 8; ++i) {
    float4 t = reinterpret_cast<const float4*>(sm.f)[tid + i * 256];
    u[i] = t;
    s1 += t.x + t.y + t.z + t.w;
    s2 += t.x * t.x + t.y * t.y + t.z * t.z + t.w * t.w;
  }
  blockReduce2(s1, s2, sm.red, tid);
  mu   = s1 * inv;
  var  = s2 * inv - mu * mu;
  rstd = rsqrtf(var + 1e-5f);
#pragma unroll
  for (int i = 0; i < 8; ++i) {
    int i4 = tid + i * 256;
    int e0 = i4 * 4;
    float4 g4  = reinterpret_cast<const float4*>(ln2g)[i4];
    float4 be4 = reinterpret_cast<const float4*>(ln2b)[i4];
    float x0 = (u[i].x - mu) * rstd * g4.x + be4.x;
    float x1 = (u[i].y - mu) * rstd * g4.y + be4.y;
    float x2 = (u[i].z - mu) * rstd * g4.z + be4.z;
    float x3 = (u[i].w - mu) * rstd * g4.w + be4.w;
    reinterpret_cast<float4*>(sm.f)[i4] = make_float4(x0, x1, x2, x3);
    int row = e0 >> 7, col = e0 & 127;
    v4bf p = { (__bf16)x0, (__bf16)x1, (__bf16)x2, (__bf16)x3 };
    *reinterpret_cast<v4bf*>(&sm.xh[row * LDH + col]) = p;  // row-major A tile
  }
  __syncthreads();

  // ---- MLP layer 1: h = GELU(x2 @ w1^T + b1) ----
#pragma unroll
  for (int mt = 0; mt < 4; ++mt) acc[mt] = vzero8();
#pragma unroll
  for (int ks = 0; ks < 4; ++ks) {  // K = 128
    v16bf bf = loadBs(sm.w1s, 8, ks, nt, lane);
#pragma unroll
    for (int mt = 0; mt < 4; ++mt) {
      v16bf af = loadA(sm.xh, LDH, mt * 16, ks * 32, lane);
      acc[mt] = wmma_bf16(af, bf, acc[mt]);
    }
  }
#pragma unroll
  for (int mt = 0; mt < 4; ++mt) {
#pragma unroll
    for (int r = 0; r < 8; ++r) {
      int tt = mt * 16 + hb * 8 + r;
      int cc = nt * 16 + nn;
      float hv = acc[mt][r] + sm.bb1[cc];
      hv = 0.5f * hv * (1.0f + erff(hv * 0.70710678118654752f));  // exact GELU
      sm.hh[tt * LDH + cc] = (__bf16)hv;
    }
  }
  __syncthreads();

  // ---- MLP layer 2 + residual: out = x2 + (h @ w2^T + b2) ----
#pragma unroll
  for (int mt = 0; mt < 4; ++mt) acc[mt] = vzero8();
#pragma unroll
  for (int ks = 0; ks < 4; ++ks) {  // K = 128
    v16bf bf = loadBs(sm.w2s, 8, ks, nt, lane);
#pragma unroll
    for (int mt = 0; mt < 4; ++mt) {
      v16bf af = loadA(sm.hh, LDH, mt * 16, ks * 32, lane);
      acc[mt] = wmma_bf16(af, bf, acc[mt]);
    }
  }
  float* gout = out + (size_t)bidx * TCn;
#pragma unroll
  for (int mt = 0; mt < 4; ++mt) {
#pragma unroll
    for (int r = 0; r < 8; ++r) {
      int tt = mt * 16 + hb * 8 + r;
      int cc = nt * 16 + nn;
      gout[tt * Cdim + cc] = acc[mt][r] + sm.bb2[cc] + sm.f[tt * Cdim + cc];
    }
  }
}

}  // namespace

extern "C" void kernel_launch(void* const* d_in, const int* in_sizes, int n_in,
                              void* d_out, int out_size, void* d_ws, size_t ws_size,
                              hipStream_t stream) {
  const float* inp  = (const float*)d_in[0];
  const float* ln1g = (const float*)d_in[1];
  const float* ln1b = (const float*)d_in[2];
  const float* ln2g = (const float*)d_in[3];
  const float* ln2b = (const float*)d_in[4];
  const float* triw = (const float*)d_in[5];
  const float* trib = (const float*)d_in[6];
  const float* w1   = (const float*)d_in[7];
  const float* b1   = (const float*)d_in[8];
  const float* w2   = (const float*)d_in[9];
  const float* b2   = (const float*)d_in[10];
  float* out = (float*)d_out;
  __bf16* wsb = (__bf16*)d_ws;

  const int Bn = in_sizes[0] / TCn;  // 4096 batch tiles, one workgroup each

  // One-shot pack: f32 weights -> bf16 swizzled image in workspace.
  prep_weights_kernel<<<dim3(16), dim3(256), 0, stream>>>(triw, w1, w2, wsb);

  (void)hipFuncSetAttribute(reinterpret_cast<const void*>(mixer2d_triu_kernel),
                            hipFuncAttributeMaxDynamicSharedMemorySize,
                            (int)sizeof(Smem));
  mixer2d_triu_kernel<<<dim3(Bn), dim3(256), sizeof(Smem), stream>>>(
      inp, ln1g, ln1b, ln2g, ln2b, triw, trib, w1, b1, w2, b2, wsb, out);
}
